// SlidingTileAttention3D_83408264888464
// MI455X (gfx1250) — compile-verified, matrix operand order checked by reading, not run on hardware
//
#include <hip/hip_runtime.h>
#include <hip/hip_bf16.h>

// ---------------------------------------------------------------------------
// Sliding-Tile 3D Attention for MI455X (gfx1250, wave32, WMMA).
// Constants baked from setup_inputs():
//   S=4096, hid=1024, H=16, d=64, NT=32 tiles of tn=128,
//   tile grid (ntt,nth,ntw)=(4,4,2), tile (tt,th,tw)=(2,8,8),
//   neighbor window = 3x2x2 = 12 kv tiles per query tile.
// Pipeline: cvt -> QKV GEMMs (f16 WMMA, 32x64 register-blocked per wave)
//           -> block-sparse flash attention (f16 WMMA + f32 softmax)
//           -> output GEMM (f32 out).
// ---------------------------------------------------------------------------

typedef __attribute__((ext_vector_type(16))) _Float16 v16h;
typedef __attribute__((ext_vector_type(8)))  _Float16 v8h;
typedef __attribute__((ext_vector_type(8)))  float    v8f;

union HV { v16h v; v8h h2[2]; };

// sequence index from (tile id nt, within-tile index q)
__device__ __forceinline__ int sidx(int nt, int q) {
    int i_ntt = nt >> 3;          // /(nth*ntw)=8
    int i_nth = (nt >> 1) & 3;
    int i_ntw = nt & 1;
    int i_tt  = q >> 6;           // /(th*tw)=64
    int i_th  = (q >> 3) & 7;
    int i_tw  = q & 7;
    // s = ((((i_ntt*2 + i_tt)*4 + i_nth)*8 + i_th)*2 + i_ntw)*8 + i_tw
    return i_ntt * 1024 + i_tt * 512 + i_nth * 128 + i_th * 16 + i_ntw * 8 + i_tw;
}

// ------------------------- conversion kernels ------------------------------

__global__ void cvt_f32_to_f16(const float* __restrict__ in,
                               _Float16* __restrict__ out, int n) {
    int i = blockIdx.x * blockDim.x + threadIdx.x;
    if (i < n) out[i] = (_Float16)in[i];
}

// W[k][n] f32 (1024x1024) -> WT[n][k] f16 so B-columns are contiguous rows
__global__ void transpose_cvt_w(const float* __restrict__ in,
                                _Float16* __restrict__ out) {
    int i = blockIdx.x * blockDim.x + threadIdx.x;   // over 1M elements
    int k = i >> 10, n = i & 1023;
    out[n * 1024 + k] = (_Float16)in[i];
}

// ------------------------------ WMMA GEMM ----------------------------------
// C[4096x1024] = A[4096x1024] (f16 row-major) x B (given as BT[1024x1024] f16,
// row n of BT = column n of B).
// Register-blocked: each wave computes a 32x64 block of C (2 m-subtiles x
// 4 n-subtiles): per k-step, 2 A-operands + 4 B-operands feed 8 WMMAs.
// grid = (4096/32, 1024/(8 waves * 64)), block = 256 (8 waves).

template <bool F16OUT>
__global__ void __launch_bounds__(256) gemm_wmma(const _Float16* __restrict__ A,
                                                 const _Float16* __restrict__ BT,
                                                 void* __restrict__ C) {
    const int lane = threadIdx.x & 31, wid = threadIdx.x >> 5;
    const int g = lane >> 4, ln = lane & 15;
    const int m0 = blockIdx.x * 32;                  // 2 m-subtiles
    const int nbase = (blockIdx.y * 8 + wid) * 64;   // 4 n-subtiles

    const _Float16* ap[2];
    ap[0] = A + (size_t)(m0 + ln) * 1024;
    ap[1] = A + (size_t)(m0 + 16 + ln) * 1024;

    v8f acc[2][4];
    v8f zero = {};
#pragma unroll
    for (int mi = 0; mi < 2; ++mi)
#pragma unroll
        for (int nt = 0; nt < 4; ++nt) acc[mi][nt] = zero;

    for (int k0 = 0; k0 < 1024; k0 += 32) {
        HV a[2];
#pragma unroll
        for (int mi = 0; mi < 2; ++mi) {
            a[mi].h2[0] = *(const v8h*)(ap[mi] + k0 + 8 * g);
            a[mi].h2[1] = *(const v8h*)(ap[mi] + k0 + 16 + 8 * g);
        }
        __builtin_prefetch(ap[0] + k0 + 128, 0, 0);
#pragma unroll
        for (int nt = 0; nt < 4; ++nt) {
            const _Float16* bp = BT + (size_t)(nbase + nt * 16 + ln) * 1024;
            HV b;
            b.h2[0] = *(const v8h*)(bp + k0 + 8 * g);
            b.h2[1] = *(const v8h*)(bp + k0 + 16 + 8 * g);
            acc[0][nt] = __builtin_amdgcn_wmma_f32_16x16x32_f16(
                false, a[0].v, false, b.v, (short)0, acc[0][nt], false, false);
            acc[1][nt] = __builtin_amdgcn_wmma_f32_16x16x32_f16(
                false, a[1].v, false, b.v, (short)0, acc[1][nt], false, false);
        }
    }
#pragma unroll
    for (int mi = 0; mi < 2; ++mi)
#pragma unroll
        for (int nt = 0; nt < 4; ++nt)
#pragma unroll
            for (int r = 0; r < 8; ++r) {
                int orow = m0 + mi * 16 + r + 8 * g;
                int ocol = nbase + nt * 16 + ln;
                if (F16OUT)
                    ((_Float16*)C)[(size_t)orow * 1024 + ocol] = (_Float16)acc[mi][nt][r];
                else
                    ((float*)C)[(size_t)orow * 1024 + ocol] = acc[mi][nt][r];
            }
}

// -------------------- block-sparse flash attention -------------------------
// grid = (NT=32, H=16), block = 256 (8 waves). Wave w owns query rows
// [w*16, w*16+16) of the 128-row query tile. Per kv neighbor tile:
// stage K (row major) + V^T into LDS, S = Q K^T via WMMA, online softmax,
// O += P V via WMMA (P staged f16 in a wave-private LDS strip).

__global__ void __launch_bounds__(256) sta3d_attn(const _Float16* __restrict__ Q,
                                                  const _Float16* __restrict__ K,
                                                  const _Float16* __restrict__ V,
                                                  _Float16* __restrict__ O) {
    __shared__ alignas(16) _Float16 Ks[128][64];    // 16 KB
    __shared__ alignas(16) _Float16 VTs[64][128];   // 16 KB
    __shared__ alignas(16) _Float16 Ps[128][128];   // 32 KB (wave-private strips)

    const int nt = blockIdx.x, h = blockIdx.y;
    const int tid = threadIdx.x;
    const int lane = tid & 31, wid = tid >> 5;
    const int g = lane >> 4, ln = lane & 15;

    // hoist this wave's Q A-operands (16 rows x 64 d) into registers
    const int qrow = wid * 16 + ln;
    const _Float16* qp = Q + (size_t)sidx(nt, qrow) * 1024 + h * 64;
    HV a0, a1;
    a0.h2[0] = *(const v8h*)(qp + 8 * g);
    a0.h2[1] = *(const v8h*)(qp + 16 + 8 * g);
    a1.h2[0] = *(const v8h*)(qp + 32 + 8 * g);
    a1.h2[1] = *(const v8h*)(qp + 48 + 8 * g);

    float mrow[8], lrow[8];
    v8f acc_o[4];
    v8f zero = {};
#pragma unroll
    for (int r = 0; r < 8; ++r) { mrow[r] = -1e30f; lrow[r] = 0.f; }
#pragma unroll
    for (int n = 0; n < 4; ++n) acc_o[n] = zero;

    // clamped sliding-window neighbor centers (ntt=4: clip to [1,2];
    // nth=4: clip to [1,3]; ntw=2: center always 1)
    const int i_ntt = nt >> 3, i_nth = (nt >> 1) & 3;
    const int ct = min(max(i_ntt, 1), 2);
    const int ch = min(max(i_nth, 1), 3);

    for (int kv = 0; kv < 12; ++kv) {
        const int jt = kv >> 2, jh = (kv >> 1) & 1, jw = kv & 1;
        const int nt_k = (ct - 1 + jt) * 8 + (ch - 1 + jh) * 2 + jw;

        __syncthreads();  // previous iteration done reading Ks/VTs
        for (int i = tid; i < 1024; i += 256) {
            const int krow = i >> 3, c8 = (i & 7) * 8;
            const size_t srow = (size_t)sidx(nt_k, krow) * 1024 + h * 64 + c8;
            *(v8h*)(&Ks[krow][c8]) = *(const v8h*)(K + srow);
            v8h vv = *(const v8h*)(V + srow);
#pragma unroll
            for (int e = 0; e < 8; ++e) VTs[c8 + e][krow] = vv[e];
        }
        __syncthreads();

        // S strip: 16 x 128, scaled by 1/sqrt(64)
        v8f sc[8];
#pragma unroll
        for (int j = 0; j < 8; ++j) {
            HV b0, b1;
            const _Float16* kk = &Ks[j * 16 + ln][0];
            b0.h2[0] = *(const v8h*)(kk + 8 * g);
            b0.h2[1] = *(const v8h*)(kk + 16 + 8 * g);
            b1.h2[0] = *(const v8h*)(kk + 32 + 8 * g);
            b1.h2[1] = *(const v8h*)(kk + 48 + 8 * g);
            v8f s = zero;
            s = __builtin_amdgcn_wmma_f32_16x16x32_f16(false, a0.v, false, b0.v,
                                                       (short)0, s, false, false);
            s = __builtin_amdgcn_wmma_f32_16x16x32_f16(false, a1.v, false, b1.v,
                                                       (short)0, s, false, false);
#pragma unroll
            for (int r = 0; r < 8; ++r) s[r] *= 0.125f;
            sc[j] = s;
        }

        // online softmax: row r (+8g) lives in one 16-lane half-group
#pragma unroll
        for (int r = 0; r < 8; ++r) {
            float v = sc[0][r];
#pragma unroll
            for (int j = 1; j < 8; ++j) v = fmaxf(v, sc[j][r]);
#pragma unroll
            for (int off = 8; off >= 1; off >>= 1)
                v = fmaxf(v, __shfl_xor(v, off, 16));
            const float mn = fmaxf(mrow[r], v);
            const float fr = __expf(mrow[r] - mn);
            mrow[r] = mn;
            float rs = 0.f;
#pragma unroll
            for (int j = 0; j < 8; ++j) {
                float p = __expf(sc[j][r] - mn);
                rs += p;
                Ps[wid * 16 + r + 8 * g][j * 16 + ln] = (_Float16)p;
            }
#pragma unroll
            for (int off = 8; off >= 1; off >>= 1)
                rs += __shfl_xor(rs, off, 16);
            lrow[r] = lrow[r] * fr + rs;
#pragma unroll
            for (int n = 0; n < 4; ++n) acc_o[n][r] *= fr;
        }

        // O += P (16x128) x V (128x64); A from wave-private Ps strip (in-order LDS)
#pragma unroll
        for (int ks = 0; ks < 4; ++ks) {
            HV pa;
            const _Float16* pp = &Ps[wid * 16 + ln][ks * 32];
            pa.h2[0] = *(const v8h*)(pp + 8 * g);
            pa.h2[1] = *(const v8h*)(pp + 16 + 8 * g);
#pragma unroll
            for (int n = 0; n < 4; ++n) {
                HV vb;
                const _Float16* vt = &VTs[n * 16 + ln][ks * 32];
                vb.h2[0] = *(const v8h*)(vt + 8 * g);
                vb.h2[1] = *(const v8h*)(vt + 16 + 8 * g);
                acc_o[n] = __builtin_amdgcn_wmma_f32_16x16x32_f16(
                    false, pa.v, false, vb.v, (short)0, acc_o[n], false, false);
            }
        }
    }

    // epilogue: normalize and scatter back to (S, hid) layout for the Wo GEMM
#pragma unroll
    for (int r = 0; r < 8; ++r) {
        const float inv = 1.f / lrow[r];
        const int q = wid * 16 + r + 8 * g;
        const size_t srow = (size_t)sidx(nt, q) * 1024 + h * 64;
#pragma unroll
        for (int n = 0; n < 4; ++n)
            O[srow + n * 16 + ln] = (_Float16)(acc_o[n][r] * inv);
    }
}

// ------------------------------- launcher ----------------------------------

extern "C" void kernel_launch(void* const* d_in, const int* in_sizes, int n_in,
                              void* d_out, int out_size, void* d_ws, size_t ws_size,
                              hipStream_t stream) {
    const float* X  = (const float*)d_in[0];
    const float* Wq = (const float*)d_in[1];
    const float* Wk = (const float*)d_in[2];
    const float* Wv = (const float*)d_in[3];
    const float* Wo = (const float*)d_in[4];
    float* out = (float*)d_out;

    // workspace partition (f16): X, 4x W^T, Q, K, V, O  -> 48 MB total
    const size_t SH = (size_t)4096 * 1024;
    const size_t WH = (size_t)1024 * 1024;
    _Float16* Xh  = (_Float16*)d_ws;
    _Float16* WqT = Xh + SH;
    _Float16* WkT = WqT + WH;
    _Float16* WvT = WkT + WH;
    _Float16* WoT = WvT + WH;
    _Float16* Qh  = WoT + WH;
    _Float16* Kh  = Qh + SH;
    _Float16* Vh  = Kh + SH;
    _Float16* Oh  = Vh + SH;

    cvt_f32_to_f16<<<(int)(SH / 256), 256, 0, stream>>>(X, Xh, (int)SH);
    transpose_cvt_w<<<(int)(WH / 256), 256, 0, stream>>>(Wq, WqT);
    transpose_cvt_w<<<(int)(WH / 256), 256, 0, stream>>>(Wk, WkT);
    transpose_cvt_w<<<(int)(WH / 256), 256, 0, stream>>>(Wv, WvT);
    transpose_cvt_w<<<(int)(WH / 256), 256, 0, stream>>>(Wo, WoT);

    dim3 ggrid(128, 2);  // 128 m-blocks x (2 groups * 8 waves * 64 cols)
    gemm_wmma<true><<<ggrid, 256, 0, stream>>>(Xh, WqT, Qh);
    gemm_wmma<true><<<ggrid, 256, 0, stream>>>(Xh, WkT, Kh);
    gemm_wmma<true><<<ggrid, 256, 0, stream>>>(Xh, WvT, Vh);

    sta3d_attn<<<dim3(32, 16), 256, 0, stream>>>(Qh, Kh, Vh, Oh);

    gemm_wmma<false><<<ggrid, 256, 0, stream>>>(Oh, WoT, out);
}
